// Fp8Linear_73967926772219
// MI455X (gfx1250) — 8-bit matrix operand order repaired and checked by reading, compiled, not run on hardware
//
#include <hip/hip_runtime.h>
#include <stdint.h>

typedef int   v16i __attribute__((ext_vector_type(16)));
typedef float v8f  __attribute__((ext_vector_type(8)));

#define FP8_MAX 448.0f

__global__ void k_init(unsigned* amax) {
    if (threadIdx.x < 2) amax[threadIdx.x] = 0u;
}

__global__ void k_amax(const float* __restrict__ in, size_t n4,
                       unsigned* __restrict__ dst) {
    __shared__ unsigned red[256];
    size_t i = (size_t)blockIdx.x * blockDim.x + threadIdx.x;
    size_t stride = (size_t)gridDim.x * blockDim.x;
    const float4* p = (const float4*)in;
    float m = 0.0f;
    for (; i < n4; i += stride) {
        float4 v = p[i];
        m = fmaxf(m, fmaxf(fmaxf(fabsf(v.x), fabsf(v.y)),
                           fmaxf(fabsf(v.z), fabsf(v.w))));
    }
    red[threadIdx.x] = __float_as_uint(m);
    __syncthreads();
    for (int s = 128; s > 0; s >>= 1) {
        if ((int)threadIdx.x < s) {
            unsigned a = red[threadIdx.x], b = red[threadIdx.x + s];
            red[threadIdx.x] = (a > b) ? a : b;
        }
        __syncthreads();
    }
    if (threadIdx.x == 0) atomicMax(dst, red[0]);
}

#if __has_builtin(__builtin_amdgcn_cvt_pk_fp8_f32)
#define HAVE_HW_FP8_CVT 1
#else
#define HAVE_HW_FP8_CVT 0
#endif

__device__ __forceinline__ unsigned char f32_to_e4m3_sw(float x) {
    unsigned u = __float_as_uint(x);
    unsigned s = (u >> 24) & 0x80u;
    u &= 0x7FFFFFFFu;
    if (u >= 0x43E00000u) return (unsigned char)(s | 0x7Eu);
    if (u < 0x3A800000u) return (unsigned char)s;
    if (u < 0x3C800000u) {
        float a = __uint_as_float(u);
        int n = (int)rintf(a * 512.0f);
        return (unsigned char)(s | (unsigned)n);
    }
    u += ((1u << 19) - 1u) + ((u >> 20) & 1u);
    unsigned e8 = ((u >> 23) & 0xFFu) - 127u + 7u;
    unsigned m3 = (u >> 20) & 7u;
    return (unsigned char)(s | (e8 << 3) | m3);
}

__device__ __forceinline__ unsigned pack4_fp8(float a, float b, float c, float d) {
#if HAVE_HW_FP8_CVT
    int v = 0;
    v = __builtin_amdgcn_cvt_pk_fp8_f32(a, b, v, false);
    v = __builtin_amdgcn_cvt_pk_fp8_f32(c, d, v, true);
    return (unsigned)v;
#else
    return (unsigned)f32_to_e4m3_sw(a) |
           ((unsigned)f32_to_e4m3_sw(b) << 8) |
           ((unsigned)f32_to_e4m3_sw(c) << 16) |
           ((unsigned)f32_to_e4m3_sw(d) << 24);
#endif
}

__global__ void k_quant(const float* __restrict__ in, uint8_t* __restrict__ outq,
                        const unsigned* __restrict__ amax, int which, size_t n4) {
    float a = __uint_as_float(amax[which]);
    float s = FP8_MAX / fmaxf(a, 1e-12f);
    size_t i = (size_t)blockIdx.x * blockDim.x + threadIdx.x;
    size_t stride = (size_t)gridDim.x * blockDim.x;
    const float4* p = (const float4*)in;
    unsigned* o = (unsigned*)outq;
    for (; i < n4; i += stride) {
        float4 v = p[i];
        float a0 = fminf(fmaxf(v.x * s, -FP8_MAX), FP8_MAX);
        float a1 = fminf(fmaxf(v.y * s, -FP8_MAX), FP8_MAX);
        float a2 = fminf(fmaxf(v.z * s, -FP8_MAX), FP8_MAX);
        float a3 = fminf(fmaxf(v.w * s, -FP8_MAX), FP8_MAX);
        o[i] = pack4_fp8(a0, a1, a2, a3);
    }
}

__device__ __forceinline__ void async_b128(const uint8_t* g, uint8_t* l) {
    unsigned loff = (unsigned)(uintptr_t)l;
    uint64_t ga   = (uint64_t)(uintptr_t)g;
    asm volatile("global_load_async_to_lds_b128 %0, %1, off"
                 :: "v"(loff), "v"(ga) : "memory");
}
__device__ __forceinline__ void wait_async0() {
    asm volatile("s_wait_asynccnt 0x0" ::: "memory");
}

#define BM 128
#define BN 128
#define BK 128
#define LDSP 144
#define TILEB (BM * LDSP)

union AFragU { v16i v; uint64_t d[8]; };
union BFragU { v16i v; uint4 q[4]; };

__global__ __launch_bounds__(256) void k_gemm(
    const uint8_t* __restrict__ Aq,
    const uint8_t* __restrict__ Bq,
    const float*   __restrict__ bias,
    const unsigned* __restrict__ amax,
    float* __restrict__ out,
    int M, int N, int K)
{
    __shared__ alignas(16) uint8_t sA[2 * TILEB];
    __shared__ alignas(16) uint8_t sB[2 * TILEB];

    const int tid  = threadIdx.x;
    const int lane = tid & 31;
    const int wave = tid >> 5;
    const int wm   = wave & 3;
    const int wn   = wave >> 2;
    const int half = lane >> 4;
    const int l16  = lane & 15;

    const int n0 = blockIdx.x * BN;
    const int m0 = blockIdx.y * BM;

    const float ax  = __uint_as_float(amax[0]);
    const float aw  = __uint_as_float(amax[1]);
    const float sx  = FP8_MAX / fmaxf(ax, 1e-12f);
    const float sw  = FP8_MAX / fmaxf(aw, 1e-12f);
    const float inv = 1.0f / (sx * sw);

    const int srow[4] = { (tid + 0 * 256) >> 3, (tid + 1 * 256) >> 3,
                          (tid + 2 * 256) >> 3, (tid + 3 * 256) >> 3 };
    const int scol = tid & 7;

    v8f acc[2][4];
    const v8f vzero = {0.f, 0.f, 0.f, 0.f, 0.f, 0.f, 0.f, 0.f};
#pragma unroll
    for (int i = 0; i < 2; ++i)
#pragma unroll
        for (int j = 0; j < 4; ++j) acc[i][j] = vzero;

    const int nK = K / BK;

    {
        const uint8_t* gA = Aq + (size_t)m0 * K;
        const uint8_t* gB = Bq + (size_t)n0 * K;
#pragma unroll
        for (int i = 0; i < 4; ++i) {
            async_b128(gA + (size_t)srow[i] * K + scol * 16,
                       sA + srow[i] * LDSP + scol * 16);
            async_b128(gB + (size_t)srow[i] * K + scol * 16,
                       sB + srow[i] * LDSP + scol * 16);
        }
    }
    wait_async0();
    __syncthreads();

    for (int ks = 0; ks < nK; ++ks) {
        const int cur = ks & 1;
        uint8_t* sAc = sA + cur * TILEB;
        uint8_t* sBc = sB + cur * TILEB;

        if (ks + 1 < nK) {
            const uint8_t* gA = Aq + (size_t)m0 * K + (size_t)(ks + 1) * BK;
            const uint8_t* gB = Bq + (size_t)n0 * K + (size_t)(ks + 1) * BK;
            uint8_t* sAn = sA + (cur ^ 1) * TILEB;
            uint8_t* sBn = sB + (cur ^ 1) * TILEB;
#pragma unroll
            for (int i = 0; i < 4; ++i) {
                async_b128(gA + (size_t)srow[i] * K + scol * 16,
                           sAn + srow[i] * LDSP + scol * 16);
                async_b128(gB + (size_t)srow[i] * K + scol * 16,
                           sBn + srow[i] * LDSP + scol * 16);
            }
        }

        AFragU afrag[2];
#pragma unroll
        for (int mt = 0; mt < 2; ++mt) {
            const uint8_t* base =
                sAc + (wm * 32 + mt * 16 + l16) * LDSP + half * 16;
#pragma unroll
            for (int p = 0; p < 8; ++p)
                afrag[mt].d[p] = *(const uint64_t*)(base + (p >> 1) * 32 + (p & 1) * 8);
        }

#pragma unroll
        for (int nt = 0; nt < 4; ++nt) {
            const uint8_t* base =
                sBc + (wn * 64 + nt * 16 + l16) * LDSP + half * 16;
            BFragU bfrag;
#pragma unroll
            for (int c = 0; c < 4; ++c)
                bfrag.q[c] = *(const uint4*)(base + c * 32);

            acc[0][nt] = __builtin_amdgcn_wmma_f32_16x16x128_fp8_fp8(
                afrag[0].v, bfrag.v, (short)0, acc[0][nt], false, false);
            acc[1][nt] = __builtin_amdgcn_wmma_f32_16x16x128_fp8_fp8(
                afrag[1].v, bfrag.v, (short)0, acc[1][nt], false, false);
        }

        wait_async0();
        __syncthreads();
    }

#pragma unroll
    for (int nt = 0; nt < 4; ++nt) {
        int gn = n0 + wn * 64 + nt * 16 + l16;
        float bv = bias[gn];
#pragma unroll
        for (int mt = 0; mt < 2; ++mt) {
            int gmBase = m0 + wm * 32 + mt * 16 + half * 8;
#pragma unroll
            for (int r = 0; r < 8; ++r) {
                out[(size_t)(gmBase + r) * N + gn] = acc[mt][nt][r] * inv + bv;
            }
        }
    }
}

extern "C" void kernel_launch(void* const* d_in, const int* in_sizes, int n_in,
                              void* d_out, int out_size, void* d_ws, size_t ws_size,
                              hipStream_t stream) {
    const float* x    = (const float*)d_in[0];
    const float* w    = (const float*)d_in[1];
    const float* bias = (const float*)d_in[2];
    float* out        = (float*)d_out;

    const int N = in_sizes[2];
    const int K = in_sizes[1] / N;
    const int M = in_sizes[0] / K;

    unsigned* amax = (unsigned*)d_ws;
    uint8_t*  xq   = (uint8_t*)d_ws + 256;
    uint8_t*  wq   = xq + (size_t)M * K;

    const size_t nx4 = ((size_t)M * K) / 4;
    const size_t nw4 = ((size_t)N * K) / 4;

    k_init<<<1, 64, 0, stream>>>(amax);
    k_amax<<<2048, 256, 0, stream>>>(x, nx4, amax + 0);
    k_amax<<<1024, 256, 0, stream>>>(w, nw4, amax + 1);
    k_quant<<<4096, 256, 0, stream>>>(x, xq, amax, 0, nx4);
    k_quant<<<2048, 256, 0, stream>>>(w, wq, amax, 1, nw4);

    dim3 grid(N / BN, M / BM);
    k_gemm<<<grid, 256, 0, stream>>>(xq, wq, bias, amax, out, M, N, K);
}
